// LSTM_32091995635996
// MI455X (gfx1250) — matrix operand order repaired and checked by reading, not compiled, not run on hardware
//
#include <hip/hip_runtime.h>

// ---------------------------------------------------------------------------
// 2-layer LSTM (B=512,T=512,F=64,H=128) + FC(10) + sigmoid, gfx1250 (MI455X).
//
//  * kernel 0: stream-convert x f32 -> f16 (one-time, ~4us at 23.3 TB/s).
//  * per layer: grid = B/BT workgroups, each owns BT=32 batch rows for the
//    whole T loop (batch rows independent -> zero cross-WG sync).
//  * per step: gates(BT x 512) = [x_t | h](BT x K) @ W^T(K x 512) via
//    v_wmma_f32_16x16x32_f16 (f32 accumulation), K = 192 (L1) / 256 (L2).
//  * 16 waves: waves 0-7 own (i,f) column slices, waves 8-15 the matching
//    (g,o) slices -> i,f,c,h stay lane-aligned in registers; only raw g/o
//    pre-activations cross waves through LDS.
//  * W^T B-fragments loaded ONCE into VGPRs (96/128 VGPRs/wave), resident
//    across all 512 steps.
//  * Tensor Data Mover does ALL per-step global traffic:
//      - tensor_load_to_lds:   input tile (BT x KIN f16, row stride T*KIN)
//        into a double-buffered LDS region, issued one step ahead; TDM
//        hardware padding (+4 DWORDs/row) gives the bank-conflict-free
//        layout.
//      - tensor_store_from_lds (L1): h1 tile from a double-buffered staging
//        region to global, issued one step behind.
//    Both share the in-order per-wave tensor queue; s_wait_tensorcnt 0 at
//    the step top fences both directions before buffers are reused.
// ---------------------------------------------------------------------------

#define Bx   512
#define Tt   512
#define Ff   64
#define Hh   128
#define NCLS 10
#define BT   32
#define NWAVES 16
#define NTHREADS (NWAVES * 32)

typedef __attribute__((ext_vector_type(16))) _Float16 v16h;
typedef __attribute__((ext_vector_type(8)))  _Float16 v8h;
typedef __attribute__((ext_vector_type(4)))  _Float16 v4h;
typedef __attribute__((ext_vector_type(8)))  float    v8f;
typedef __attribute__((ext_vector_type(4)))  unsigned int u32x4;
typedef __attribute__((ext_vector_type(8)))  int          i32x8;
typedef __attribute__((ext_vector_type(4)))  int          i32x4;

__device__ __forceinline__ float fsig(float x) {
  return 1.0f / (1.0f + __expf(-x));
}
__device__ __forceinline__ float ftanh(float x) {
  float ax = __builtin_fabsf(x);
  float e  = __expf(-2.0f * ax);
  float r  = (1.0f - e) / (1.0f + e);
  return __builtin_copysignf(r, x);
}

// Low 32 bits of the flat shared-aperture address are the LDS byte offset
// (CDNA5 aperture mapping).
__device__ __forceinline__ unsigned int lds_offset(const void* p) {
  return (unsigned int)(unsigned long long)(uintptr_t)p;
}

// TDM 2D tile transfer: ROWS x DIM0 f16 elements, global row stride
// row_stride elements. PAD (loads only): +4 DWORDs LDS padding per row
// (row = DIM0/2 DWORDs -> padded stride = DIM0+8 halves).
// Descriptor bitfields per CDNA5 ISA ch.8 (D# groups 0/1); 2D tensor ->
// groups 2/3 unused (zero).
template <int DIM0, int ROWS, bool PAD, bool STORE>
__device__ __forceinline__ void tdm_tile(unsigned int lds_off,
                                         const void* gptr, int row_stride) {
  const unsigned long long ga = (unsigned long long)(uintptr_t)gptr;
  u32x4 g0;
  g0[0] = 1u;                                   // count=1 (valid user D#)
  g0[1] = lds_off;                              // lds_addr [63:32]
  g0[2] = (unsigned int)ga;                     // global_addr [95:64]
  g0[3] = (unsigned int)((ga >> 32) & 0x1FFFFFFu) | (2u << 30); // [120:96]|type=2

  constexpr unsigned PADI = (DIM0 / 2 == 32) ? 4u : 5u;  // pad every 32/64 DW
  constexpr unsigned PADBITS =
      PAD ? ((1u << 20) | (PADI << 22) | (3u << 25)) : 0u;
  i32x8 g1;
  g1[0] = (int)((1u << 16) | PADBITS);   // data_size = 1 -> 2 bytes (+pad)
  g1[1] = (int)((unsigned)DIM0 << 16);   // tensor_dim0[79:48]
  g1[2] = (int)((unsigned)ROWS << 16);   // tensor_dim1[111:80]
  g1[3] = (int)((unsigned)DIM0 << 16);   // tile_dim0[127:112]
  g1[4] = ROWS;                          // tile_dim1[143:128]
  g1[5] = row_stride;                    // tensor_dim0_stride[207:160]
  g1[6] = 0;
  g1[7] = 0;
  const i32x4 z4 = {0, 0, 0, 0};
  const i32x8 z8 = {0, 0, 0, 0, 0, 0, 0, 0};
  if constexpr (STORE)
    __builtin_amdgcn_tensor_store_from_lds(g0, g1, z4, z4, z8, 0);
  else
    __builtin_amdgcn_tensor_load_to_lds(g0, g1, z4, z4, z8, 0);
}

// A-fragment (16x32 f16): lanes 0-15 hold K[0..16) of row M=lane,
// lanes 16-31 hold K[16..32) of row M=lane-16 -> 2x ds_load_b128.
__device__ __forceinline__ v16h load_afrag(const _Float16* p) {
  v8h lo = *(const v8h*)p;
  v8h hi = *(const v8h*)(p + 8);
  return __builtin_shufflevector(lo, hi, 0, 1, 2, 3, 4, 5, 6, 7,
                                          8, 9, 10, 11, 12, 13, 14, 15);
}

// B-fragment (32x16 f16): lane<16 -> K[0..16) of col N=lane, lane>=16 ->
// K[16..32) of col N=lane-16. Weights (4H x K) row-major -> contiguous read.
template <int KIH>
__device__ __forceinline__ v16h load_bfrag(const float* __restrict__ Wih,
                                           const float* __restrict__ Whh,
                                           int n, int kbase) {
  v16h r;
#pragma unroll
  for (int kk = 0; kk < 16; ++kk) {
    int k = kbase + kk;
    float v = (k < KIH) ? Wih[n * KIH + k] : Whh[n * Hh + (k - KIH)];
    r[kk] = (_Float16)v;
  }
  return r;
}

__global__ __launch_bounds__(256) void cvt_f32_to_f16(
    const float* __restrict__ in, _Float16* __restrict__ out, int n4) {
  int i = blockIdx.x * blockDim.x + threadIdx.x;
  const int stride = gridDim.x * blockDim.x;
  for (; i < n4; i += stride) {
    const float4 v = ((const float4*)in)[i];
    v4h h = { (_Float16)v.x, (_Float16)v.y, (_Float16)v.z, (_Float16)v.w };
    ((v4h*)out)[i] = h;
  }
}

// KIN: 64 (layer1) or 128 (layer2). L1: stream h1 to hout via TDM stores.
// DO_FC: fused FC+sigmoid epilogue (layer2).
template <int KIN, bool L1, bool DO_FC>
__global__ __launch_bounds__(NTHREADS, 1) void lstm_layer_kernel(
    const _Float16* __restrict__ xin,   // (B,T,KIN) f16
    const float*    __restrict__ Wih,   // (4H, KIN)
    const float*    __restrict__ Whh,   // (4H, H)
    const float*    __restrict__ bih,   // (4H)
    const float*    __restrict__ bhh,   // (4H)
    _Float16*       __restrict__ hout,  // (B,T,H) f16   (L1 only)
    const float*    __restrict__ Wfc,   // (NCLS, H)     (DO_FC only)
    const float*    __restrict__ bfc,   // (NCLS)        (DO_FC only)
    float*          __restrict__ out)   // (B, NCLS)     (DO_FC only)
{
  constexpr int KTOT   = KIN + Hh;
  constexpr int KSTEPS = KTOT / 32;
  constexpr int XSTEPS = KIN / 32;
  constexpr int XST    = KIN + 8;   // padded stride (matches TDM padding)
  constexpr int HST    = Hh + 8;

  __shared__ __align__(16) _Float16 xbuf[2][BT][XST];  // TDM-load double buffer
  __shared__ __align__(16) _Float16 hbuf[BT][HST];     // recurrent h (f16)
  __shared__ __align__(16) float    g_buf[BT][Hh + 4];
  __shared__ __align__(16) float    o_buf[BT][Hh + 4];
  // TDM-store staging for h1 (unpadded: store reads LDS contiguously)
  __shared__ __align__(16) _Float16 sbuf[L1 ? 2 : 1][BT][L1 ? Hh : 8];

  const int tid  = threadIdx.x;
  const int lane = tid & 31;
  const int wv   = tid >> 5;
  const int half = lane >> 4;
  const int lidx = lane & 15;
  const int b0   = blockIdx.x * BT;

  // Gate-column ownership: waves 0-7 -> (i,f), waves 8-15 -> (g,o); pair
  // (w, w+8) covers hidden units [16w, 16w+16).
  const bool ifw  = (wv < 8);
  const int  wsub = ifw ? wv : (wv - 8);
  const int  n0   = (ifw ? 0   : 256) + 16 * wsub + lidx;
  const int  n1   = (ifw ? 128 : 384) + 16 * wsub + lidx;

  // Resident weight fragments (loaded once, reused 512x).
  v16h wB0[KSTEPS], wB1[KSTEPS];
#pragma unroll
  for (int kt = 0; kt < KSTEPS; ++kt) {
    const int kb = kt * 32 + half * 16;
    wB0[kt] = load_bfrag<KIN>(Wih, Whh, n0, kb);
    wB1[kt] = load_bfrag<KIN>(Wih, Whh, n1, kb);
  }
  const float bias0 = bih[n0] + bhh[n0];
  const float bias1 = bih[n1] + bhh[n1];

  float cst[2][8];
#pragma unroll
  for (int m = 0; m < 2; ++m)
#pragma unroll
    for (int r = 0; r < 8; ++r) cst[m][r] = 0.0f;

  for (int i = tid; i < BT * Hh; i += NTHREADS)
    hbuf[i / Hh][i % Hh] = (_Float16)0.0f;

  // Prologue: DMA tile t=0 into xbuf[0] (wave 0 owns the TDM queue).
  if (wv == 0)
    tdm_tile<KIN, BT, true, false>(lds_offset(&xbuf[0][0][0]),
                                   xin + (size_t)b0 * Tt * KIN, Tt * KIN);

  for (int t = 0; t < Tt; ++t) {
    const int p = t & 1;
    // Fence the tensor queue: load-t landed, and (L1) store-(t-2) drained,
    // so sbuf[p] / xbuf[p] are safe to reuse.
    if (wv == 0) __builtin_amdgcn_s_wait_tensorcnt(0);
    __syncthreads();  // publishes xbuf[p] + h/sbuf from previous step

    if (wv == 0) {
      if constexpr (L1) {
        if (t > 0)  // DMA h1(t-1) out of sbuf[1-p] while this step computes
          tdm_tile<Hh, BT, false, true>(
              lds_offset(&sbuf[1 - p][0][0]),
              hout + ((size_t)b0 * Tt + (t - 1)) * Hh, Tt * Hh);
      }
      if (t + 1 < Tt)  // DMA next input tile into the other buffer
        tdm_tile<KIN, BT, true, false>(
            lds_offset(&xbuf[1 - p][0][0]),
            xin + ((size_t)b0 * Tt + (t + 1)) * KIN, Tt * KIN);
    }

    // gates = [x_t | h] @ W^T (+bias via accumulator init)
    v8f acc0[2], acc1[2];
#pragma unroll
    for (int m = 0; m < 2; ++m)
#pragma unroll
      for (int r = 0; r < 8; ++r) { acc0[m][r] = bias0; acc1[m][r] = bias1; }

#pragma unroll
    for (int kt = 0; kt < KSTEPS; ++kt) {
      const bool fromX = (kt < XSTEPS);
      const int  ko = (fromX ? kt * 32 : kt * 32 - KIN) + half * 16;
      const _Float16* r0 = fromX ? &xbuf[p][lidx][ko]      : &hbuf[lidx][ko];
      const _Float16* r1 = fromX ? &xbuf[p][16 + lidx][ko] : &hbuf[16 + lidx][ko];
      const v16h aF0 = load_afrag(r0);
      const v16h aF1 = load_afrag(r1);
      acc0[0] = __builtin_amdgcn_wmma_f32_16x16x32_f16(
          false, aF0, false, wB0[kt], (short)0, acc0[0], false, false);
      acc0[1] = __builtin_amdgcn_wmma_f32_16x16x32_f16(
          false, aF1, false, wB0[kt], (short)0, acc0[1], false, false);
      acc1[0] = __builtin_amdgcn_wmma_f32_16x16x32_f16(
          false, aF0, false, wB1[kt], (short)0, acc1[0], false, false);
      acc1[1] = __builtin_amdgcn_wmma_f32_16x16x32_f16(
          false, aF1, false, wB1[kt], (short)0, acc1[1], false, false);
    }

    // waves 8-15: export raw g/o pre-activations
    if (!ifw) {
#pragma unroll
      for (int m = 0; m < 2; ++m)
#pragma unroll
        for (int r = 0; r < 8; ++r) {
          const int row = m * 16 + r + 8 * half;
          const int col = 16 * wsub + lidx;
          g_buf[row][col] = acc0[m][r];
          o_buf[row][col] = acc1[m][r];
        }
    }
    __syncthreads();  // g/o visible to (i,f) waves

    // waves 0-7: LSTM cell update, write new h
    if (ifw) {
#pragma unroll
      for (int m = 0; m < 2; ++m)
#pragma unroll
        for (int r = 0; r < 8; ++r) {
          const int row = m * 16 + r + 8 * half;
          const int col = 16 * wsub + lidx;
          const float iv = fsig(acc0[m][r]);
          const float fv = fsig(acc1[m][r]);
          const float gv = ftanh(g_buf[row][col]);
          const float ov = fsig(o_buf[row][col]);
          const float c  = fv * cst[m][r] + iv * gv;
          cst[m][r] = c;
          const float h = ov * ftanh(c);
          hbuf[row][col] = (_Float16)h;
          if constexpr (L1)
            sbuf[p][row][col] = (_Float16)h;   // staged for TDM store
        }
    }
    // next iteration's top barrier orders hbuf/sbuf writes before use;
    // TDM traffic targets buffers nobody touches this step.
  }

  if constexpr (L1) {
    __syncthreads();  // all of sbuf[(T-1)&1] written
    if (wv == 0)
      tdm_tile<Hh, BT, false, true>(
          lds_offset(&sbuf[(Tt - 1) & 1][0][0]),
          hout + ((size_t)b0 * Tt + (Tt - 1)) * Hh, Tt * Hh);
    // s_endpgm performs an implicit wait-idle (all counters, incl. TENSORcnt)
  }

  if constexpr (DO_FC) {
    __syncthreads();  // final h2 resident in hbuf
    if (tid < BT * NCLS) {
      const int row = tid / NCLS;
      const int cls = tid % NCLS;
      float s = bfc[cls];
#pragma unroll 8
      for (int k = 0; k < Hh; ++k)
        s += (float)hbuf[row][k] * Wfc[cls * Hh + k];
      out[(size_t)(b0 + row) * NCLS + cls] = fsig(s);
    }
  }
}

extern "C" void kernel_launch(void* const* d_in, const int* in_sizes, int n_in,
                              void* d_out, int out_size, void* d_ws,
                              size_t ws_size, hipStream_t stream) {
  const float* x     = (const float*)d_in[0];
  const float* w1_ih = (const float*)d_in[1];
  const float* w1_hh = (const float*)d_in[2];
  const float* b1_ih = (const float*)d_in[3];
  const float* b1_hh = (const float*)d_in[4];
  const float* w2_ih = (const float*)d_in[5];
  const float* w2_hh = (const float*)d_in[6];
  const float* b2_ih = (const float*)d_in[7];
  const float* b2_hh = (const float*)d_in[8];
  const float* w_fc  = (const float*)d_in[9];
  const float* b_fc  = (const float*)d_in[10];

  // workspace: h1 (B,T,H) f16 = 64 MB, then x16 (B,T,F) f16 = 32 MB
  _Float16* h1  = (_Float16*)d_ws;
  _Float16* x16 = (_Float16*)((char*)d_ws + (size_t)Bx * Tt * Hh * 2);
  float*    out = (float*)d_out;

  cvt_f32_to_f16<<<4096, 256, 0, stream>>>(x, x16, Bx * Tt * Ff / 4);

  lstm_layer_kernel<Ff, true, false><<<Bx / BT, NTHREADS, 0, stream>>>(
      x16, w1_ih, w1_hh, b1_ih, b1_hh, h1, nullptr, nullptr, nullptr);

  lstm_layer_kernel<Hh, false, true><<<Bx / BT, NTHREADS, 0, stream>>>(
      h1, w2_ih, w2_hh, b2_ih, b2_hh, nullptr, w_fc, b_fc, out);
}
